// biLSTM_CNN_CRF_16939351015425
// MI455X (gfx1250) — compile-verified, matrix operand order checked by reading, not run on hardware
//
#include <hip/hip_runtime.h>
#include <hip/hip_bf16.h>

typedef _Float16 half_t;
typedef _Float16 v4h  __attribute__((ext_vector_type(4)));
typedef _Float16 v8h  __attribute__((ext_vector_type(8)));
typedef _Float16 v16h __attribute__((ext_vector_type(16)));
typedef float    v4f  __attribute__((ext_vector_type(4)));
typedef float    v8f  __attribute__((ext_vector_type(8)));

#define S_LEN 512
#define B_SZ  64
#define E_DIM 256
#define H_DIM 256
#define G_DIM 1024   // 4*H
#define ROWS  (S_LEN * B_SZ)   // 32768

__device__ __forceinline__ v16h cat8(v8h lo, v8h hi) {
  return __builtin_shufflevector(lo, hi, 0,1,2,3,4,5,6,7,8,9,10,11,12,13,14,15);
}

// ---------------- f32 -> f16 conversion ----------------
__global__ __launch_bounds__(256) void k_cvt_f16(const float* __restrict__ src,
                                                 half_t* __restrict__ dst, int n) {
  int i = blockIdx.x * 256 + threadIdx.x;
  if (i < n) dst[i] = (half_t)src[i];
}

// ---------------- embedding gather -> f16, layout row = s*64+b ----------------
__global__ __launch_bounds__(256) void k_embed(const int* __restrict__ sentence, // [B][S]
                                               const float* __restrict__ emb,    // [V][E]
                                               half_t* __restrict__ X) {         // [ROWS][E]
  int row = blockIdx.x;            // s*64 + b
  int s = row >> 6, b = row & 63;
  int tok = sentence[b * S_LEN + s];
  int e = threadIdx.x;
  X[(size_t)row * E_DIM + e] = (half_t)emb[(size_t)tok * E_DIM + e];
}

// ---------------- big input-projection GEMM: Xp = X @ Wih^T (f16 in/out, f32 acc) ----
// X: [ROWS][256], W: [1024][256] row-major (row n over k), Xp: [ROWS][1024]
__global__ __launch_bounds__(256) void k_gemm_inproj(const half_t* __restrict__ X,
                                                     const half_t* __restrict__ W,
                                                     half_t* __restrict__ Xp) {
  int wave = threadIdx.x >> 5;
  int lane = threadIdx.x & 31;
  int lm = lane & 15, hi = lane >> 4;
  int job = blockIdx.x * 8 + wave;       // 32768 jobs
  int mt  = job >> 4;                    // 0..2047
  int nt0 = (job & 15) * 4;              // n-tile group of 4

  v8f acc[4] = {};
  const half_t* arow = X + (size_t)(mt * 16 + lm) * E_DIM;
  #pragma unroll
  for (int kk = 0; kk < 8; ++kk) {
    int k0 = kk * 32 + (hi << 3);
    v16h a = cat8(*(const v8h*)(arow + k0), *(const v8h*)(arow + k0 + 16));
    #pragma unroll
    for (int j = 0; j < 4; ++j) {
      const half_t* brow = W + (size_t)((nt0 + j) * 16 + lm) * E_DIM;
      v16h bf = cat8(*(const v8h*)(brow + k0), *(const v8h*)(brow + k0 + 16));
      acc[j] = __builtin_amdgcn_wmma_f32_16x16x32_f16(false, a, false, bf,
                                                      (short)0, acc[j], false, false);
    }
  }
  #pragma unroll
  for (int j = 0; j < 4; ++j) {
    int n = (nt0 + j) * 16 + lm;
    #pragma unroll
    for (int r = 0; r < 8; ++r) {
      int m = mt * 16 + 8 * hi + r;
      Xp[(size_t)m * G_DIM + n] = (half_t)acc[j][r];
    }
  }
}

// ---------------- persistent bidirectional LSTM recurrence, batch-split ----------------
// grid = 8 blocks: dir = blk>>2, batch group = blk&3 (16 rows each = one WMMA M-tile).
// Whh fragments held REGISTER-RESIDENT across all 512 steps (2 n-tiles x 8 k = 128 VGPR).
// LDS: g (16x1024 f32 = 64KB) + h (16x256 f16 = 8KB).
__global__ __launch_bounds__(1024) void k_lstm(const half_t* __restrict__ XpF,
                                               const half_t* __restrict__ XpB,
                                               const half_t* __restrict__ WhhF,
                                               const half_t* __restrict__ WhhB,
                                               const float* __restrict__ bF,
                                               const float* __restrict__ bB,
                                               float* __restrict__ hs) {   // [ROWS][512]
  extern __shared__ char smem[];
  float*  g_lds = (float*)smem;                              // 16*1024
  half_t* h_lds = (half_t*)(smem + 16 * G_DIM * 4);          // 16*256

  int dir = blockIdx.x >> 2;
  int b0  = (blockIdx.x & 3) * 16;                           // first batch row of group
  const half_t* Xp   = dir ? XpB  : XpF;
  const half_t* Whh  = dir ? WhhB : WhhF;
  const float*  bias = dir ? bB : bF;
  int col_off = dir * H_DIM;

  int tid = threadIdx.x;
  int lane = tid & 31, wave = tid >> 5;
  int lm = lane & 15, hi = lane >> 4;
  int bl = tid >> 6;               // local batch row (0..15) for elementwise
  int j0 = (tid * 4) & 255;        // 4 hidden units per thread

  // ---- preload register-resident Whh fragments ----
  v16h Bf[2][8];
  #pragma unroll
  for (int nj = 0; nj < 2; ++nj) {
    const half_t* brow = Whh + (size_t)((wave * 2 + nj) * 16 + lm) * H_DIM + (hi << 3);
    #pragma unroll
    for (int kk = 0; kk < 8; ++kk)
      Bf[nj][kk] = cat8(*(const v8h*)(brow + kk * 32), *(const v8h*)(brow + kk * 32 + 16));
  }

  // ---- preload per-thread bias (constant across steps) ----
  v4f bi = *(const v4f*)(bias + j0);
  v4f bfg = *(const v4f*)(bias + 256 + j0);
  v4f bg = *(const v4f*)(bias + 512 + j0);
  v4f bo = *(const v4f*)(bias + 768 + j0);

  for (int i = tid; i < 16 * H_DIM; i += 1024) h_lds[i] = (half_t)0.f;
  float c[4] = {0.f, 0.f, 0.f, 0.f};
  __syncthreads();

  #pragma unroll 1
  for (int t = 0; t < S_LEN; ++t) {
    int s = dir ? (S_LEN - 1 - t) : t;

    // ---- GEMM: g = h_prev @ Whh^T (wave: 1 m-tile x 2 n-tiles, K=256) ----
    v8f acc[2] = {};
    #pragma unroll
    for (int kk = 0; kk < 8; ++kk) {
      int k0 = kk * 32 + (hi << 3);
      const half_t* hr = &h_lds[lm * H_DIM + k0];
      v16h a = cat8(*(const v8h*)hr, *(const v8h*)(hr + 16));
      acc[0] = __builtin_amdgcn_wmma_f32_16x16x32_f16(false, a, false, Bf[0][kk],
                                                      (short)0, acc[0], false, false);
      acc[1] = __builtin_amdgcn_wmma_f32_16x16x32_f16(false, a, false, Bf[1][kk],
                                                      (short)0, acc[1], false, false);
    }
    #pragma unroll
    for (int nj = 0; nj < 2; ++nj) {
      int n = (wave * 2 + nj) * 16 + lm;
      #pragma unroll
      for (int r = 0; r < 8; ++r)
        g_lds[(8 * hi + r) * G_DIM + n] = acc[nj][r];
    }
    __syncthreads();

    // ---- elementwise LSTM cell: thread = (local row bl, 4 hidden units j0..j0+3) ----
    const half_t* xp_row = Xp + (size_t)(s * B_SZ + b0 + bl) * G_DIM;
    if (t + 1 < S_LEN) {
      int sn = dir ? (s - 1) : (s + 1);
      __builtin_prefetch(Xp + (size_t)(sn * B_SZ + b0 + bl) * G_DIM + j0, 0, 0);
    }
    v4f gi4 = *(v4f*)&g_lds[bl * G_DIM + j0];
    v4f gf4 = *(v4f*)&g_lds[bl * G_DIM + 256 + j0];
    v4f gg4 = *(v4f*)&g_lds[bl * G_DIM + 512 + j0];
    v4f go4 = *(v4f*)&g_lds[bl * G_DIM + 768 + j0];
    v4h xi4 = *(const v4h*)(xp_row + j0);
    v4h xf4 = *(const v4h*)(xp_row + 256 + j0);
    v4h xg4 = *(const v4h*)(xp_row + 512 + j0);
    v4h xo4 = *(const v4h*)(xp_row + 768 + j0);
    v4f hn4;
    v4h hh4;
    #pragma unroll
    for (int p = 0; p < 4; ++p) {
      float gi = gi4[p] + (float)xi4[p] + bi[p];
      float gf = gf4[p] + (float)xf4[p] + bfg[p];
      float gg = gg4[p] + (float)xg4[p] + bg[p];
      float go = go4[p] + (float)xo4[p] + bo[p];
      float si = 1.f / (1.f + __expf(-gi));
      float sf = 1.f / (1.f + __expf(-gf));
      float so = 1.f / (1.f + __expf(-go));
      c[p] = sf * c[p] + si * tanhf(gg);
      float hn = so * tanhf(c[p]);
      hn4[p] = hn;
      hh4[p] = (half_t)hn;
    }
    *(v4h*)&h_lds[bl * H_DIM + j0] = hh4;
    *(v4f*)&hs[(size_t)(s * B_SZ + b0 + bl) * (2 * H_DIM) + col_off + j0] = hn4;
    __syncthreads();
  }
}

// ---------------- emissions: emis = [hf|hb] @ Wlin^T + blin (T=4: plain f32) ----
__global__ __launch_bounds__(256) void k_linear(const float* __restrict__ hs,
                                                const float* __restrict__ Wlin, // [4][512]
                                                const float* __restrict__ blin,
                                                float* __restrict__ emis) {     // [ROWS][4]
  int g = blockIdx.x * 256 + threadIdx.x;   // ROWS*4
  int row = g >> 2, tg = g & 3;
  const float4* h4 = (const float4*)(hs + (size_t)row * 512);
  const float4* w4 = (const float4*)(Wlin + (size_t)tg * 512);
  float acc = blin[tg];
  #pragma unroll 4
  for (int i = 0; i < 128; ++i) {
    float4 a = h4[i], b = w4[i];
    acc += a.x * b.x + a.y * b.y + a.z * b.z + a.w * b.w;
  }
  emis[g] = acc;
}

// ---------------- CRF forward (loss) + Viterbi decode, one thread per batch ----
__global__ __launch_bounds__(64) void k_crf(const float* __restrict__ emis,  // [S][B][4]
                                            const int* __restrict__ tags,    // [B][S]
                                            const float* __restrict__ st,
                                            const float* __restrict__ en,
                                            const float* __restrict__ trans, // [4][4]
                                            int* __restrict__ bp,            // [S-1][B][4]
                                            float* __restrict__ out) {       // 32769
  int b = threadIdx.x;
  float tr[4][4], s0[4], e0[4];
  #pragma unroll
  for (int i = 0; i < 4; ++i) {
    s0[i] = st[i]; e0[i] = en[i];
    #pragma unroll
    for (int j = 0; j < 4; ++j) tr[i][j] = trans[i * 4 + j];
  }
  float alpha[4], score[4];
  const float* e = emis + (size_t)b * 4;
  #pragma unroll
  for (int i = 0; i < 4; ++i) { alpha[i] = s0[i] + e[i]; score[i] = alpha[i]; }
  int tprev = tags[b * S_LEN];
  float num = s0[tprev] + e[tprev];

  for (int s = 1; s < S_LEN; ++s) {
    e = emis + ((size_t)s * B_SZ + b) * 4;
    float na[4], ns[4];
    #pragma unroll
    for (int tgt = 0; tgt < 4; ++tgt) {
      float m = alpha[0] + tr[0][tgt];
      #pragma unroll
      for (int k = 1; k < 4; ++k) { float v = alpha[k] + tr[k][tgt]; m = v > m ? v : m; }
      float ssum = 0.f;
      #pragma unroll
      for (int k = 0; k < 4; ++k) ssum += __expf(alpha[k] + tr[k][tgt] - m);
      na[tgt] = m + __logf(ssum) + e[tgt];

      float best = score[0] + tr[0][tgt]; int bi = 0;
      #pragma unroll
      for (int k = 1; k < 4; ++k) {
        float v = score[k] + tr[k][tgt];
        if (v > best) { best = v; bi = k; }
      }
      ns[tgt] = best + e[tgt];
      bp[(((size_t)(s - 1)) * B_SZ + b) * 4 + tgt] = bi;
    }
    #pragma unroll
    for (int i = 0; i < 4; ++i) { alpha[i] = na[i]; score[i] = ns[i]; }
    int tg = tags[b * S_LEN + s];
    num += tr[tprev][tg] + e[tg];
    tprev = tg;
  }
  num += e0[tprev];

  float m = alpha[0] + e0[0];
  #pragma unroll
  for (int i = 1; i < 4; ++i) { float v = alpha[i] + e0[i]; m = v > m ? v : m; }
  float ssum = 0.f;
  #pragma unroll
  for (int i = 0; i < 4; ++i) ssum += __expf(alpha[i] + e0[i] - m);
  float denom = m + __logf(ssum);

  float best = score[0] + e0[0]; int last = 0;
  #pragma unroll
  for (int i = 1; i < 4; ++i) {
    float v = score[i] + e0[i];
    if (v > best) { best = v; last = i; }
  }
  out[b * S_LEN + (S_LEN - 1)] = (float)last;
  int tg = last;
  for (int s = S_LEN - 2; s >= 0; --s) {
    tg = bp[((size_t)s * B_SZ + b) * 4 + tg];
    out[b * S_LEN + s] = (float)tg;
  }

  __shared__ float red[64];
  red[b] = num - denom;
  __syncthreads();
  if (b == 0) {
    float acc = 0.f;
    for (int i = 0; i < 64; ++i) acc += red[i];
    out[B_SZ * S_LEN] = -acc;   // loss = -sum(llh)
  }
}

extern "C" void kernel_launch(void* const* d_in, const int* in_sizes, int n_in,
                              void* d_out, int out_size, void* d_ws, size_t ws_size,
                              hipStream_t stream) {
  const int*   sentence = (const int*)d_in[0];
  const int*   tags     = (const int*)d_in[1];
  const float* emb      = (const float*)d_in[3];
  const float* Wih_f    = (const float*)d_in[4];
  const float* Whh_f    = (const float*)d_in[5];
  const float* b_f      = (const float*)d_in[6];
  const float* Wih_b    = (const float*)d_in[7];
  const float* Whh_b    = (const float*)d_in[8];
  const float* b_b      = (const float*)d_in[9];
  const float* Wlin     = (const float*)d_in[10];
  const float* blin     = (const float*)d_in[11];
  const float* start_t  = (const float*)d_in[12];
  const float* end_t    = (const float*)d_in[13];
  const float* trans    = (const float*)d_in[14];
  float* out = (float*)d_out;

  char* base = (char*)d_ws;
  size_t off = 0;
  auto alloc = [&](size_t bytes) -> char* {
    char* p = base + off;
    off += (bytes + 255) & ~(size_t)255;
    return p;
  };
  half_t* Xh   = (half_t*)alloc((size_t)ROWS * E_DIM * 2);      // 16 MB
  half_t* WihF = (half_t*)alloc((size_t)G_DIM * E_DIM * 2);
  half_t* WhhF = (half_t*)alloc((size_t)G_DIM * H_DIM * 2);
  half_t* WihB = (half_t*)alloc((size_t)G_DIM * E_DIM * 2);
  half_t* WhhB = (half_t*)alloc((size_t)G_DIM * H_DIM * 2);
  half_t* XpF  = (half_t*)alloc((size_t)ROWS * G_DIM * 2);      // 67 MB
  half_t* XpB  = (half_t*)alloc((size_t)ROWS * G_DIM * 2);      // 67 MB
  float*  hsb  = (float*)alloc((size_t)ROWS * 512 * 4);         // 67 MB
  float*  emis = (float*)alloc((size_t)ROWS * 4 * 4);
  int*    bp   = (int*)alloc((size_t)(S_LEN - 1) * B_SZ * 4 * 4);

  const int WN = G_DIM * E_DIM;   // 262144
  k_cvt_f16<<<WN / 256, 256, 0, stream>>>(Wih_f, WihF, WN);
  k_cvt_f16<<<WN / 256, 256, 0, stream>>>(Whh_f, WhhF, WN);
  k_cvt_f16<<<WN / 256, 256, 0, stream>>>(Wih_b, WihB, WN);
  k_cvt_f16<<<WN / 256, 256, 0, stream>>>(Whh_b, WhhB, WN);

  k_embed<<<ROWS, 256, 0, stream>>>(sentence, emb, Xh);

  k_gemm_inproj<<<4096, 256, 0, stream>>>(Xh, WihF, XpF);
  k_gemm_inproj<<<4096, 256, 0, stream>>>(Xh, WihB, XpB);

  // 8 persistent recurrence blocks: 2 dirs x 4 batch groups of 16 rows
  size_t shmem = (size_t)16 * G_DIM * 4 + (size_t)16 * H_DIM * 2;  // 72 KB
  k_lstm<<<8, 1024, shmem, stream>>>(XpF, XpB, WhhF, WhhB, b_f, b_b, hsb);

  k_linear<<<(ROWS * 4) / 256, 256, 0, stream>>>(hsb, Wlin, blin, emis);

  k_crf<<<1, 64, 0, stream>>>(emis, tags, start_t, end_t, trans, bp, out);
}
